// SparseWindowAttention_40973988004145
// MI455X (gfx1250) — matrix operand order repaired and checked by reading, not compiled, not run on hardware
//
#include <hip/hip_runtime.h>

// ---------------- types ----------------
typedef _Float16 h16;
typedef __attribute__((ext_vector_type(16))) _Float16 v16h;
typedef __attribute__((ext_vector_type(8)))  float    v8f;

// CDNA5 async global->LDS path (ASYNCcnt), guarded so we always compile.
#if __has_builtin(__builtin_amdgcn_global_load_async_to_lds_b128) && \
    __has_builtin(__builtin_amdgcn_s_wait_asynccnt)
#define USE_ASYNC_LDS 1
#else
#define USE_ASYNC_LDS 0
#endif

#if USE_ASYNC_LDS
// builtin signature (from hipcc diagnostic): param0 = int4* in global AS(1),
// param1 (by symmetry) = int4* in LDS AS(3), then imm offset, imm cpol.
typedef int v4i __attribute__((vector_size(16)));
typedef __attribute__((address_space(1))) v4i* gas4_p;
typedef __attribute__((address_space(3))) v4i* las4_p;
#define ASYNC_B128(gsrc, ldst) \
  __builtin_amdgcn_global_load_async_to_lds_b128((gas4_p)(gsrc), (las4_p)(ldst), 0, 0)
#endif

// ---------------- static problem config ----------------
#define T_    4
#define H_    48
#define W_    48
#define C_    512
#define NH_   8
#define CH_   64
#define NWH_  6
#define NWW_  6
#define NW_   36
#define WA_   64
#define P_    144       // pooled tokens per frame (12x12)
#define NKF_  400       // kv tokens per frame: 64 + 192 + 144
#define MX_   (T_*H_*W_) // 9216 spatial tokens

union Frag { v16h v; uint4 u[2]; h16 h[16]; };

// A fragment (16x32 f16): lane holds row (lane&15); halves
// [kb..kb+7] and [kb+16..kb+23] with kb = k0 + (hi-lane ? 8 : 0)
__device__ __forceinline__ v16h load_a(const h16* row, int k0, int lane) {
  Frag f;
  const h16* p = row + k0 + ((lane & 16) ? 8 : 0);
  f.u[0] = *(const uint4*)(p);
  f.u[1] = *(const uint4*)(p + 16);
  return f.v;
}
// B fragment (32x16 f16): lane holds column (lane&15); halves
// [kb..kb+7] and [kb+16..kb+23] with kb = k0 + (hi-lane ? 8 : 0)
__device__ __forceinline__ v16h load_b(const h16* row, int k0, int lane) {
  Frag f;
  const h16* p = row + k0 + ((lane & 16) ? 8 : 0);
  f.u[0] = *(const uint4*)(p);
  f.u[1] = *(const uint4*)(p + 16);
  return f.v;
}

// ---------------- prep kernels ----------------
__global__ __launch_bounds__(256) void k_f32_to_f16(const float* __restrict__ s,
                                                    h16* __restrict__ d, int n) {
  int i = blockIdx.x * 256 + threadIdx.x;
  if (i < n) d[i] = (h16)s[i];
}

// WT[n][k] = W[k][n], f32 -> f16 (512x512)
__global__ __launch_bounds__(256) void k_transpose_w(const float* __restrict__ Wsrc,
                                                     h16* __restrict__ WT) {
  int i = blockIdx.x * 256 + threadIdx.x;   // i = n*512 + k
  int n = i >> 9, k = i & 511;
  WT[i] = (h16)Wsrc[k * 512 + n];
}

// depthwise 4x4 stride-4 pool: px[t][p][c] f16
__global__ __launch_bounds__(256) void k_pool(const float* __restrict__ x,
                                              const float* __restrict__ pw,
                                              const float* __restrict__ pb,
                                              h16* __restrict__ px) {
  int i = blockIdx.x * 256 + threadIdx.x;   // over T_*P_*C_
  if (i >= T_ * P_ * C_) return;
  int c  = i & (C_ - 1);
  int tp = i >> 9;
  int t  = tp / P_, p = tp - t * P_;
  int ph = p / 12, pwid = p - ph * 12;
  float s = 0.f;
  for (int a = 0; a < 4; ++a)
    for (int b = 0; b < 4; ++b) {
      float xv = x[((size_t)((t * H_ + ph * 4 + a) * W_ + pwid * 4 + b)) * C_ + c];
      s += xv * pw[c * 16 + a * 4 + b];
    }
  px[i] = (h16)(s + pb[c]);
}

// per-window masked flag
__global__ void k_wm(const float* __restrict__ mask, int* __restrict__ wm) {
  int win = threadIdx.x;
  if (win >= NW_) return;
  int wy = win / NWW_, wx = win % NWW_;
  int f = 0;
  for (int t = 0; t < T_; ++t)
    for (int i = 0; i < 8; ++i)
      for (int j = 0; j < 8; ++j)
        if (mask[((t * H_ + wy * 8 + i) * W_ + wx * 8 + j)] > 0.f) f = 1;
  wm[win] = f;
}

// reconstruct VALID_IND (192 entries) on device
__global__ void k_vind(int* __restrict__ vind) {
  if (threadIdx.x != 0 || blockIdx.x != 0) return;
  int cnt = 0;
  for (int s = 0; s < 256; ++s) {
    int g = s >> 6, lo = s & 63, i = lo >> 3, j = lo & 7;
    bool invalid;
    if (g == 0)      invalid = (i < 4)  && (j < 4);
    else if (g == 1) invalid = (i < 4)  && (j >= 4);
    else if (g == 2) invalid = (i >= 4) && (j < 4);
    else             invalid = (i >= 4) && (j >= 4);
    if (!invalid) vind[cnt++] = s;
  }
}

// ---------------- WMMA GEMM: Out[M,N] = A[M,K] * BT[N,K]^T + bias ----------------
// one wave per 16x64 output tile; 16|M, 64|N, 32|K
template <typename OutT>
__global__ __launch_bounds__(256) void gemm_bias(const h16* __restrict__ A,
                                                 const h16* __restrict__ BT,
                                                 const float* __restrict__ bias,
                                                 OutT* __restrict__ Out,
                                                 int M, int N, int K) {
  int wid  = (blockIdx.x << 3) + (threadIdx.x >> 5);
  int lane = threadIdx.x & 31;
  int nblk = N >> 6;
  int mt = wid / nblk, nt = wid % nblk;
  if (mt * 16 >= M) return;

  v8f acc[4] = {};
  const h16* arow = A + (size_t)(mt * 16 + (lane & 15)) * K;
  const h16* brow = BT + (size_t)(nt * 64 + (lane & 15)) * K;

  for (int k0 = 0; k0 < K; k0 += 32) {
    v16h a = load_a(arow, k0, lane);
#pragma unroll
    for (int j = 0; j < 4; ++j) {
      v16h b = load_b(brow + (size_t)(j * 16) * K, k0, lane);
      acc[j] = __builtin_amdgcn_wmma_f32_16x16x32_f16(false, a, false, b,
                                                      (short)0, acc[j], false, false);
    }
  }

  int mbase = mt * 16 + ((lane & 16) ? 8 : 0);
#pragma unroll
  for (int j = 0; j < 4; ++j) {
    int n = nt * 64 + j * 16 + (lane & 15);
    float bn = bias[n];
#pragma unroll
    for (int r = 0; r < 8; ++r)
      Out[(size_t)(mbase + r) * N + n] = (OutT)(acc[j][r] + bn);
  }
}

// ---------------- flash attention per (window, head) ----------------
__global__ __launch_bounds__(256) void attn_kernel(
    const h16* __restrict__ qh, const h16* __restrict__ kh, const h16* __restrict__ vh,
    const h16* __restrict__ pkh, const h16* __restrict__ pvh,
    const int* __restrict__ wm, const int* __restrict__ vind,
    h16* __restrict__ yh) {
  __shared__ h16 Qs[256][64];      // 32 KB
  __shared__ h16 Kc[64][64];       //  8 KB (token-major)
  __shared__ h16 Vt[64][64];       //  8 KB (channel-major)
  __shared__ h16 Pw[8][16][64];    // 16 KB per-wave probs

  const int win = blockIdx.x / NH_, head = blockIdx.x % NH_;
  const int wy = win / NWW_, wx = win % NWW_;
  const int tid = threadIdx.x, lane = tid & 31, wave = tid >> 5;
  const bool maskedwin = wm[win] != 0;

  // stage Q: one token per thread (async global->LDS when available)
  {
    int q = tid;
    int tq = q >> 6, lo = q & 63, i = lo >> 3, j = lo & 7;
    const h16* src = qh + ((size_t)((tq * H_ + wy * 8 + i) * W_ + wx * 8 + j)) * C_ + head * CH_;
#if USE_ASYNC_LDS
#pragma unroll
    for (int e = 0; e < 8; ++e)
      ASYNC_B128(src + e * 8, &Qs[q][e * 8]);
#else
    uint4* d = (uint4*)Qs[q];
    const uint4* s = (const uint4*)src;
#pragma unroll
    for (int e = 0; e < 8; ++e) d[e] = s[e];
#endif
  }

  v8f acc[4] = {};
  float mrow[8], lrow[8];
#pragma unroll
  for (int r = 0; r < 8; ++r) { mrow[r] = -1e30f; lrow[r] = 0.f; }

  const int qframe = wave >> 2;   // 16 q-rows per wave, 64 per frame
  const float scale = 0.125f;     // 1/sqrt(64)

  for (int c = 0; c < 25; ++c) {  // 25 * 64 = 1600 kv tokens
    __syncthreads();
    // gather 64 kv tokens: 4 threads per token, 16 halves each
    {
      int tk = tid >> 2, part = tid & 3;
      int kk = c * 64 + tk;
      int tt = kk / NKF_, j = kk - tt * NKF_;
      const h16 *ks, *vs;
      if (j < 64) {
        int i = j >> 3, jj = j & 7;
        size_t off = ((size_t)((tt * H_ + wy * 8 + i) * W_ + wx * 8 + jj)) * C_ + head * CH_;
        ks = kh + off; vs = vh + off;
      } else if (j < 256) {
        int s = vind[j - 64];
        int g = s >> 6, lo = s & 63, i = lo >> 3, jj = lo & 7;
        int sy = (g < 2) ? -4 : 4;
        int sx = (g & 1) ? 4 : -4;
        int hs = (wy * 8 + i - sy + H_) % H_;
        int wsrc = (wx * 8 + jj - sx + W_) % W_;
        size_t off = ((size_t)((tt * H_ + hs) * W_ + wsrc)) * C_ + head * CH_;
        ks = kh + off; vs = vh + off;
      } else {
        int p = j - 256;
        size_t off = ((size_t)(tt * P_ + p)) * C_ + head * CH_;
        ks = pkh + off; vs = pvh + off;
      }
      // K tile, token-major: two b128 per thread
#if USE_ASYNC_LDS
      ASYNC_B128(ks + part * 16, &Kc[tk][part * 16]);
      ASYNC_B128(ks + part * 16 + 8, &Kc[tk][part * 16 + 8]);
#else
      {
        uint4* d = (uint4*)&Kc[tk][part * 16];
        const uint4* s4 = (const uint4*)(ks + part * 16);
        d[0] = s4[0]; d[1] = s4[1];
      }
#endif
      // V tile, channel-major (transpose while storing)
      {
        const h16* vsp = vs + part * 16;
#pragma unroll
        for (int e = 0; e < 16; ++e) Vt[part * 16 + e][tk] = vsp[e];
      }
    }
#if USE_ASYNC_LDS
    __builtin_amdgcn_s_wait_asynccnt(0);
#endif
    __syncthreads();

    // logits: 4 N-tiles of 16, K=64 channels in 2 WMMA steps
    float S[4][8];
#pragma unroll
    for (int nt = 0; nt < 4; ++nt) {
      v8f s8 = {};
#pragma unroll
      for (int kk2 = 0; kk2 < 2; ++kk2) {
        v16h a = load_a(Qs[wave * 16 + (lane & 15)], kk2 * 32, lane);
        v16h b = load_b(Kc[nt * 16 + (lane & 15)], kk2 * 32, lane);
        s8 = __builtin_amdgcn_wmma_f32_16x16x32_f16(false, a, false, b,
                                                    (short)0, s8, false, false);
      }
      int kv = c * 64 + nt * 16 + (lane & 15);
      int kvf = kv / NKF_, kj = kv - kvf * NKF_;
      bool allowed = maskedwin || ((kvf == qframe) && (kj < 64));
      float bias = allowed ? 0.f : -1e9f;
#pragma unroll
      for (int r = 0; r < 8; ++r) S[nt][r] = s8[r] * scale + bias;
    }

    // online softmax per row-slot (row lives across a 16-lane half)
#pragma unroll
    for (int r = 0; r < 8; ++r) {
      float cm = fmaxf(fmaxf(S[0][r], S[1][r]), fmaxf(S[2][r], S[3][r]));
#pragma unroll
      for (int m = 1; m < 16; m <<= 1) cm = fmaxf(cm, __shfl_xor(cm, m, 16));
      float mnew  = fmaxf(mrow[r], cm);
      float alpha = __expf(mrow[r] - mnew);
      float rs = 0.f;
#pragma unroll
      for (int nt = 0; nt < 4; ++nt) {
        float p = __expf(S[nt][r] - mnew);
        S[nt][r] = p;
        rs += p;
      }
#pragma unroll
      for (int m = 1; m < 16; m <<= 1) rs += __shfl_xor(rs, m, 16);
      lrow[r] = lrow[r] * alpha + rs;
      mrow[r] = mnew;
#pragma unroll
      for (int ct = 0; ct < 4; ++ct) acc[ct][r] *= alpha;
    }

    // re-fragment probs through per-wave LDS (same-wave DS is in-order)
    {
      int rofs = (lane & 16) ? 8 : 0;
#pragma unroll
      for (int nt = 0; nt < 4; ++nt)
#pragma unroll
        for (int r = 0; r < 8; ++r)
          Pw[wave][rofs + r][nt * 16 + (lane & 15)] = (h16)S[nt][r];
    }

    // acc += P (16x64) @ V (64x64), V channel-major as B
#pragma unroll
    for (int ct = 0; ct < 4; ++ct) {
#pragma unroll
      for (int kk2 = 0; kk2 < 2; ++kk2) {
        v16h a = load_a(Pw[wave][lane & 15], kk2 * 32, lane);
        v16h b = load_b(Vt[ct * 16 + (lane & 15)], kk2 * 32, lane);
        acc[ct] = __builtin_amdgcn_wmma_f32_16x16x32_f16(false, a, false, b,
                                                         (short)0, acc[ct], false, false);
      }
    }
  }

  // normalize and scatter back to (t,H,W,C) as f16 for the output projection
  {
    int rofs = (lane & 16) ? 8 : 0;
#pragma unroll
    for (int ct = 0; ct < 4; ++ct) {
      int ch = ct * 16 + (lane & 15);
#pragma unroll
      for (int r = 0; r < 8; ++r) {
        int q = wave * 16 + rofs + r;
        int tq = q >> 6, lo = q & 63, i = lo >> 3, j = lo & 7;
        float val = acc[ct][r] / lrow[r];
        yh[((size_t)((tq * H_ + wy * 8 + i) * W_ + wx * 8 + j)) * C_ + head * CH_ + ch] = (h16)val;
      }
    }
  }
}

// ---------------- host orchestration ----------------
extern "C" void kernel_launch(void* const* d_in, const int* in_sizes, int n_in,
                              void* d_out, int out_size, void* d_ws, size_t ws_size,
                              hipStream_t stream) {
  (void)in_sizes; (void)n_in; (void)out_size; (void)ws_size;
  const float* x    = (const float*)d_in[0];
  const float* mask = (const float*)d_in[1];
  const float* Wq   = (const float*)d_in[2];
  const float* bq   = (const float*)d_in[3];
  const float* Wk   = (const float*)d_in[4];
  const float* bk   = (const float*)d_in[5];
  const float* Wv   = (const float*)d_in[6];
  const float* bv   = (const float*)d_in[7];
  const float* Wp   = (const float*)d_in[8];
  const float* bp   = (const float*)d_in[9];
  const float* pw   = (const float*)d_in[10];
  const float* pb   = (const float*)d_in[11];

  char* ws = (char*)d_ws;
  size_t off = 0;
  auto alloc = [&](size_t bytes) -> void* {
    void* p = ws + off;
    off = (off + bytes + 255) & ~(size_t)255;
    return p;
  };
  h16* xh  = (h16*)alloc((size_t)MX_ * C_ * 2);
  h16* wqT = (h16*)alloc((size_t)512 * 512 * 2);
  h16* wkT = (h16*)alloc((size_t)512 * 512 * 2);
  h16* wvT = (h16*)alloc((size_t)512 * 512 * 2);
  h16* wpT = (h16*)alloc((size_t)512 * 512 * 2);
  h16* qh  = (h16*)alloc((size_t)MX_ * C_ * 2);
  h16* kh  = (h16*)alloc((size_t)MX_ * C_ * 2);
  h16* vh  = (h16*)alloc((size_t)MX_ * C_ * 2);
  h16* pxh = (h16*)alloc((size_t)T_ * P_ * C_ * 2);
  h16* pkh = (h16*)alloc((size_t)T_ * P_ * C_ * 2);
  h16* pvh = (h16*)alloc((size_t)T_ * P_ * C_ * 2);
  h16* yh  = (h16*)alloc((size_t)MX_ * C_ * 2);
  int* wm   = (int*)alloc(NW_ * sizeof(int));
  int* vind = (int*)alloc(192 * sizeof(int));

  // prep
  {
    int n = MX_ * C_;
    k_f32_to_f16<<<(n + 255) / 256, 256, 0, stream>>>(x, xh, n);
  }
  k_transpose_w<<<1024, 256, 0, stream>>>(Wq, wqT);
  k_transpose_w<<<1024, 256, 0, stream>>>(Wk, wkT);
  k_transpose_w<<<1024, 256, 0, stream>>>(Wv, wvT);
  k_transpose_w<<<1024, 256, 0, stream>>>(Wp, wpT);
  {
    int n = T_ * P_ * C_;
    k_pool<<<(n + 255) / 256, 256, 0, stream>>>(x, pw, pb, pxh);
  }
  k_wm<<<1, 64, 0, stream>>>(mask, wm);
  k_vind<<<1, 32, 0, stream>>>(vind);

  // QKV projections (f16 out): waves = (M/16)*(N/64), 8 waves/block
  {
    int blocks = (MX_ / 16) * (C_ / 64) / 8;   // 576
    gemm_bias<h16><<<blocks, 256, 0, stream>>>(xh, wqT, bq, qh, MX_, C_, C_);
    gemm_bias<h16><<<blocks, 256, 0, stream>>>(xh, wkT, bk, kh, MX_, C_, C_);
    gemm_bias<h16><<<blocks, 256, 0, stream>>>(xh, wvT, bv, vh, MX_, C_, C_);
  }
  // pooled K/V projections
  {
    int Mp = T_ * P_;                          // 576
    int blocks = (Mp / 16) * (C_ / 64) / 8;    // 36
    gemm_bias<h16><<<blocks, 256, 0, stream>>>(pxh, wkT, bk, pkh, Mp, C_, C_);
    gemm_bias<h16><<<blocks, 256, 0, stream>>>(pxh, wvT, bv, pvh, Mp, C_, C_);
  }

  // attention: one block per (window, head)
  attn_kernel<<<NW_ * NH_, 256, 0, stream>>>(qh, kh, vh, pkh, pvh, wm, vind, yh);

  // output projection (f32 out)
  {
    int blocks = (MX_ / 16) * (C_ / 64) / 8;   // 576
    gemm_bias<float><<<blocks, 256, 0, stream>>>(yh, wpT, bp, (float*)d_out, MX_, C_, C_);
  }
}
